// EncoderRNN_42941083025634
// MI455X (gfx1250) — matrix operand order repaired and checked by reading, not compiled, not run on hardware
//
#include <hip/hip_runtime.h>

#define TLEN 512
#define BSZ  64
#define HDIM 1024
#define VOC  32000

typedef __bf16 bf16_t;
typedef __attribute__((ext_vector_type(16))) __bf16 v16bf;
typedef __attribute__((ext_vector_type(8)))  __bf16 bf16x8;
typedef __attribute__((ext_vector_type(8)))  float  v8f;

struct PtrPack {
  const float* w[8];   // fwd i,o,f,g then bwd i,o,f,g  (each [H][2H] row-major)
  const float* b[8];   // matching biases [H]
};

// ---------------------------------------------------------------------------
// Embedding gather + bf16 convert: x[t][b][h] = emb_w[h][tok] + emb_b[h]
// ---------------------------------------------------------------------------
__global__ void embed_kernel(const int* __restrict__ tokens,
                             const float* __restrict__ emb_w,
                             const float* __restrict__ emb_b,
                             bf16_t* __restrict__ x) {
  size_t idx = (size_t)blockIdx.x * blockDim.x + threadIdx.x;  // < T*B*H
  int h = (int)(idx & (HDIM - 1));
  size_t tb = idx >> 10;               // t*B + b
  int tok = tokens[tb];
  float v = emb_w[(size_t)h * VOC + (size_t)tok] + emb_b[h];
  x[idx] = (bf16_t)v;
}

// ---------------------------------------------------------------------------
// Weight fp32 -> bf16 (layout preserved: [mat][n][k], k in [0,2H))
// ---------------------------------------------------------------------------
__global__ void wconv_kernel(PtrPack pp, bf16_t* __restrict__ wbf) {
  size_t idx = (size_t)blockIdx.x * blockDim.x + threadIdx.x;  // < 8*H*2H
  int m = (int)(idx >> 21);                                    // H*2H = 2^21
  size_t r = idx & (((size_t)1 << 21) - 1);
  wbf[idx] = (bf16_t)pp.w[m][r];
}

__global__ void bconv_kernel(PtrPack pp, float* __restrict__ biasp) {
  int idx = blockIdx.x * blockDim.x + threadIdx.x;             // < 8*H
  biasp[idx] = pp.b[idx >> 10][idx & (HDIM - 1)];
}

// zero cell state (2*B*H f32) and both h recurrence buffers (2*2*B*H bf16)
__global__ void init_kernel(float* __restrict__ cst, bf16_t* __restrict__ hbuf) {
  int idx = blockIdx.x * blockDim.x + threadIdx.x;             // < 2*2*B*H
  if (idx < 2 * BSZ * HDIM) cst[idx] = 0.0f;
  hbuf[idx] = (bf16_t)0.0f;
}

// ---------------------------------------------------------------------------
// load 16 bf16 for one WMMA operand: two 16-byte chunks at p and p+sep
// (A and B layouts: sep=16 elements)
// ---------------------------------------------------------------------------
__device__ __forceinline__ v16bf load_pair(const bf16_t* __restrict__ p, int sep) {
  bf16x8 lo = *reinterpret_cast<const bf16x8*>(p);
  bf16x8 hi = *reinterpret_cast<const bf16x8*>(p + sep);
  v16bf r;
#pragma unroll
  for (int i = 0; i < 8; ++i) { r[i] = lo[i]; r[i + 8] = hi[i]; }
  return r;
}

__device__ __forceinline__ v8f wmma_bf16(v16bf a, v16bf b, v8f acc) {
  return __builtin_amdgcn_wmma_f32_16x16x32_bf16(
      /*neg_a=*/false, a, /*neg_b=*/false, b,
      /*c_mod=*/(short)0, acc, /*reuse_a=*/false, /*reuse_b=*/false);
}

// ---------------------------------------------------------------------------
// One LSTM timestep, both directions.
// grid: 128 blocks = dir(2) x hblock(64); block: 256 threads (8 waves).
// wave (gate, mhalf) computes TWO 16x16 f32 tiles (batch rows m0..m0+31)
// sharing one B tile per kc: 6 x b128 loads per 2 x v_wmma (3 loads/wmma).
// Gates combine through LDS, then the LSTM cell update is applied.
// ---------------------------------------------------------------------------
__launch_bounds__(256)
__global__ void lstm_step_kernel(const bf16_t* __restrict__ x,      // [T][B][H] bf16
                                 const bf16_t* __restrict__ wbf,    // [8][H][2H] bf16
                                 const float*  __restrict__ biasp,  // [8][H]
                                 const bf16_t* __restrict__ hprev,  // [2][B][H] bf16
                                 bf16_t*       __restrict__ hnext,  // [2][B][H] bf16
                                 float*        __restrict__ cstate, // [2][B][H] f32
                                 float*        __restrict__ out_h,  // [T][B][2H]
                                 float*        __restrict__ out_c,  // [T][B][2H]
                                 int s) {
  __shared__ float lds[4][64][17];   // [gate][batch row][n] + pad

  const int tid   = threadIdx.x;
  const int lane  = tid & 31;
  const int wave  = tid >> 5;       // 0..7
  const int gate  = wave >> 1;      // 0:i 1:o 2:f 3:g
  const int mhalf = wave & 1;       // which 32-row batch half

  const int blk  = blockIdx.x;      // 0..127
  const int dir  = blk >> 6;        // 0 fwd, 1 bwd
  const int hblk = blk & 63;        // 0..63

  const int n0   = hblk * 16;
  const int m0   = mhalf * 32;      // rows m0..m0+31 (tiles m0 and m0+16)
  const int col  = lane & 15;
  const int hi16 = lane >> 4;       // 0 or 1

  const int t_in = (dir == 0) ? s : (TLEN - 1 - s);

  // accumulators initialized with gate bias (constant along this lane's N col)
  const float bias = biasp[(dir * 4 + gate) * HDIM + n0 + col];
  v8f acc0, acc1;
#pragma unroll
  for (int v = 0; v < 8; ++v) { acc0[v] = bias; acc1[v] = bias; }

  // A lane addressing (16-bit A 16x32 layout): lane<16 holds K {0..7,16..23},
  // lane>=16 holds K {8..15,24..31} of each 32-wide chunk; row = base+(lane&15)
  const int aoff = hi16 * 8;
  const bf16_t* aX0 = x     + ((size_t)t_in * BSZ + m0 +      col) * HDIM + aoff;
  const bf16_t* aX1 = x     + ((size_t)t_in * BSZ + m0 + 16 + col) * HDIM + aoff;
  const bf16_t* aH0 = hprev + ((size_t)dir  * BSZ + m0 +      col) * HDIM + aoff;
  const bf16_t* aH1 = hprev + ((size_t)dir  * BSZ + m0 + 16 + col) * HDIM + aoff;

  // B lane addressing (bf16 B 32x16): lane holds W row n0+col; lane<16 holds
  // K {0..7,16..23}, lanes>=16 hold K {8..15,24..31} of each 32-wide chunk.
  const bf16_t* bW = wbf
      + ((size_t)(dir * 4 + gate) * HDIM + (size_t)(n0 + col)) * (2 * HDIM)
      + hi16 * 8;

  // K phase 1: x contribution (k = 0..1023)
#pragma unroll 4
  for (int kc = 0; kc < 32; ++kc) {
    v16bf b  = load_pair(bW + kc * 32, 16);
    v16bf a0 = load_pair(aX0 + kc * 32, 16);
    v16bf a1 = load_pair(aX1 + kc * 32, 16);
    acc0 = wmma_bf16(a0, b, acc0);
    acc1 = wmma_bf16(a1, b, acc1);
  }
  // K phase 2: h contribution (k = 1024..2047)
#pragma unroll 4
  for (int kc = 0; kc < 32; ++kc) {
    v16bf b  = load_pair(bW + (kc + 32) * 32, 16);
    v16bf a0 = load_pair(aH0 + kc * 32, 16);
    v16bf a1 = load_pair(aH1 + kc * 32, 16);
    acc0 = wmma_bf16(a0, b, acc0);
    acc1 = wmma_bf16(a1, b, acc1);
  }

  // scatter C tiles to LDS: VGPR v -> row v (+8 for upper lane half), col = N
#pragma unroll
  for (int v = 0; v < 8; ++v) {
    lds[gate][m0 +      v + hi16 * 8][col] = acc0[v];
    lds[gate][m0 + 16 + v + hi16 * 8][col] = acc1[v];
  }
  __syncthreads();

  // pointwise LSTM cell update: 64x16 (b,h) cells, 4 per thread
  for (int cell = tid; cell < BSZ * 16; cell += 256) {
    int b    = cell >> 4;
    int c0   = cell & 15;
    int hidx = n0 + c0;
    float iv = lds[0][b][c0];
    float ov = lds[1][b][c0];
    float fv = lds[2][b][c0];
    float gv = lds[3][b][c0];
    float ig = 1.0f / (1.0f + __expf(-iv));
    float og = 1.0f / (1.0f + __expf(-ov));
    float fg = 1.0f / (1.0f + __expf(-fv));
    float cg = tanhf(gv);
    size_t cidx = ((size_t)dir * BSZ + b) * HDIM + hidx;
    float cn = fg * cstate[cidx] + ig * cg;
    float hn = og * tanhf(cn);
    cstate[cidx] = cn;
    hnext[cidx]  = (bf16_t)hn;
    size_t oidx = ((size_t)s * BSZ + b) * (2 * HDIM) + (size_t)dir * HDIM + hidx;
    out_h[oidx] = hn;
    out_c[oidx] = cn;
  }
}

// ---------------------------------------------------------------------------
extern "C" void kernel_launch(void* const* d_in, const int* in_sizes, int n_in,
                              void* d_out, int out_size, void* d_ws, size_t ws_size,
                              hipStream_t stream) {
  (void)in_sizes; (void)n_in; (void)out_size; (void)ws_size;

  const int*   tokens = (const int*)d_in[0];
  const float* emb_w  = (const float*)d_in[1];
  const float* emb_b  = (const float*)d_in[2];

  PtrPack pp;
  for (int g = 0; g < 8; ++g) {
    pp.w[g] = (const float*)d_in[3 + 2 * g];
    pp.b[g] = (const float*)d_in[4 + 2 * g];
  }

  // workspace carve-up (~102 MB total)
  char* ws = (char*)d_ws;
  size_t off = 0;
  auto carve = [&](size_t bytes) -> void* {
    void* p = ws + off;
    off = (off + bytes + 255) & ~(size_t)255;
    return p;
  };
  bf16_t* x     = (bf16_t*)carve((size_t)TLEN * BSZ * HDIM * sizeof(bf16_t));     // 64 MB
  bf16_t* wbf   = (bf16_t*)carve((size_t)8 * HDIM * 2 * HDIM * sizeof(bf16_t));   // 33.5 MB
  float*  biasp = (float*) carve((size_t)8 * HDIM * sizeof(float));
  float*  cst   = (float*) carve((size_t)2 * BSZ * HDIM * sizeof(float));
  bf16_t* hbuf  = (bf16_t*)carve((size_t)2 * 2 * BSZ * HDIM * sizeof(bf16_t));    // double-buffered h

  embed_kernel<<<(TLEN * BSZ * HDIM) / 256, 256, 0, stream>>>(tokens, emb_w, emb_b, x);
  wconv_kernel<<<(8 * HDIM * 2 * HDIM) / 256, 256, 0, stream>>>(pp, wbf);
  bconv_kernel<<<(8 * HDIM) / 256, 256, 0, stream>>>(pp, biasp);
  init_kernel<<<(2 * 2 * BSZ * HDIM) / 256, 256, 0, stream>>>(cst, hbuf);

  float* out_h = (float*)d_out;
  float* out_c = out_h + (size_t)TLEN * BSZ * 2 * HDIM;

  const size_t hstride = (size_t)2 * BSZ * HDIM;
  for (int s = 0; s < TLEN; ++s) {
    const bf16_t* hp = hbuf + (size_t)(s & 1) * hstride;
    bf16_t*       hn = hbuf + (size_t)((s + 1) & 1) * hstride;
    lstm_step_kernel<<<128, 256, 0, stream>>>(x, wbf, biasp, hp, hn, cst,
                                              out_h, out_c, s);
  }
}
